// GRUEncoder_49615462203959
// MI455X (gfx1250) — matrix operand order repaired and checked by reading, not compiled, not run on hardware
//
#include <hip/hip_runtime.h>
#include <hip/hip_bf16.h>
#include <stdint.h>

// ---------------------------------------------------------------------------
// CDNA5 (gfx1250, wave32) dilated-GRU encoder.
// Per-layer persistent kernel: one block (2 waves) owns 16 recurrence chains.
// Per step:  gi = x @ Wih^T + bih   (v_wmma_f32_16x16x32_bf16, or VALU for I==1)
//            gh = h @ Whh^T + bhh   (v_wmma_f32_16x16x32_bf16)
// A-fragments hoisted into registers per step; fragments loaded as b128.
// N-tiles split across the 2 waves with compile-time trip counts (NT even).
// ---------------------------------------------------------------------------

typedef __bf16 bf16_t;
typedef bf16_t v16bf __attribute__((ext_vector_type(16)));
typedef float  v8f   __attribute__((ext_vector_type(8)));

union FragBF { uint4 q[2]; unsigned int u[8]; v16bf v; };

__device__ __forceinline__ unsigned short f32_to_bf16_rne(float f) {
  union { float f; unsigned int u; } x; x.f = f;
  unsigned int r = (x.u + 0x7FFFu + ((x.u >> 16) & 1u)) >> 16;
  return (unsigned short)r;
}
__device__ __forceinline__ float fast_sigmoid(float x) { return 1.0f / (1.0f + __expf(-x)); }
__device__ __forceinline__ float fast_tanh(float x) {
  float e = __expf(-2.0f * x);
  return (1.0f - e) / (1.0f + e);
}

#define FLAG_REV  2   // read raw data time-reversed
#define FLAG_LAST 4   // only compute chains with j == dil-1 (last-output-only layer)

// --- weight prep: f32 -> padded bf16 (zero-filled padding), biases padded f32
__global__ void prep_weights_kernel(const float* __restrict__ wih, const float* __restrict__ whh,
                                    const float* __restrict__ bih, const float* __restrict__ bhh,
                                    unsigned short* __restrict__ wih_p, unsigned short* __restrict__ whh_p,
                                    float* __restrict__ bih_p, float* __restrict__ bhh_p,
                                    float* __restrict__ wcol_p,
                                    int threeH, int I, int Ipad, int H, int Hpad, int NT16)
{
  int tid = blockIdx.x * blockDim.x + threadIdx.x;
  int stride = gridDim.x * blockDim.x;
  for (int idx = tid; idx < NT16 * Ipad; idx += stride) {
    int n = idx / Ipad, k = idx - n * Ipad;
    float v = (n < threeH && k < I) ? wih[n * I + k] : 0.0f;
    wih_p[idx] = f32_to_bf16_rne(v);
  }
  for (int idx = tid; idx < NT16 * Hpad; idx += stride) {
    int n = idx / Hpad, k = idx - n * Hpad;
    float v = (n < threeH && k < H) ? whh[n * H + k] : 0.0f;
    whh_p[idx] = f32_to_bf16_rne(v);
  }
  for (int n = tid; n < NT16; n += stride) {
    bih_p[n]  = (n < threeH) ? bih[n] : 0.0f;
    bhh_p[n]  = (n < threeH) ? bhh[n] : 0.0f;
    wcol_p[n] = (n < threeH) ? wih[n * I] : 0.0f;   // I==1 column for raw layers
  }
}

__global__ void zero_bytes_kernel(uint4* __restrict__ p, long long n16) {
  long long tid = blockIdx.x * (long long)blockDim.x + threadIdx.x;
  long long stride = gridDim.x * (long long)blockDim.x;
  uint4 z = {0u, 0u, 0u, 0u};
  for (long long i = tid; i < n16; i += stride) p[i] = z;
}

// --- persistent dilated-GRU layer: 2 waves per 16-chain tile, N-tiles split
template<int H, int Hpad, int IpadX, int IpadOut, int NT, int KC_gi, int KC_gh, bool RAW>
__global__ __launch_bounds__(64)
void gru_layer(const void* __restrict__ xin, unsigned short* __restrict__ yout,
               const unsigned short* __restrict__ wih_p, const unsigned short* __restrict__ whh_p,
               const float* __restrict__ bih_p, const float* __restrict__ bhh_p,
               const float* __restrict__ wcol_p,
               float* __restrict__ dout, int out_off,
               int S, int dil, int nc, int flags)
{
  static_assert((NT & 1) == 0, "NT must be even: N-tiles split across 2 waves");
  constexpr int NT16 = NT * 16;
  extern __shared__ char smem[];
  char* p = smem;
  int*   s_rowb = (int*)p;   p += 16 * 4;
  int*   s_bidx = (int*)p;   p += 16 * 4;
  int*   s_wout = (int*)p;   p += 16 * 4;
  float* s_xrow = (float*)p; p += 16 * 4;
  float* s_bih  = (float*)p; p += (size_t)NT16 * 4;
  float* s_bhh  = (float*)p; p += (size_t)NT16 * 4;
  float* s_wcol = (float*)p; p += (size_t)NT16 * 4;
  float* s_ggi  = (float*)p; p += (size_t)16 * NT16 * 4;
  float* s_ggh  = (float*)p; p += (size_t)16 * NT16 * 4;
  float* s_h    = (float*)p; p += (size_t)16 * Hpad * 4;
  unsigned short* s_hbf = (unsigned short*)p; p += (size_t)16 * Hpad * 2;
  unsigned short* s_whh = (unsigned short*)p; p += (size_t)NT16 * Hpad * 2;
  unsigned short* s_wih = (unsigned short*)p;

  const int tid  = threadIdx.x;     // 0..63 (2 waves)
  const int lane = tid & 31;
  const int wv   = tid >> 5;        // wave id: N-tile partition
  const int M    = lane & 15;
  const int hi   = lane >> 4;       // 0: low half, 1: high half of wave

  // stage weights/biases into LDS (once), zero hidden state
  for (int i = tid; i < NT16 * Hpad; i += 64) s_whh[i] = whh_p[i];
  if (!RAW)
    for (int i = tid; i < NT16 * IpadX; i += 64) s_wih[i] = wih_p[i];
  for (int i = tid; i < NT16; i += 64) { s_bih[i] = bih_p[i]; s_bhh[i] = bhh_p[i]; s_wcol[i] = wcol_p[i]; }
  for (int i = tid; i < 16 * Hpad; i += 64) { s_h[i] = 0.0f; s_hbf[i] = 0; }

  if (tid < 16) {
    int cc = blockIdx.x * 16 + tid;                  // chain index
    if (flags & FLAG_LAST) cc = cc * dil + (dil - 1);
    int b = cc / dil, j = cc - b * dil;
    s_rowb[tid] = b * S + j;                         // row index at t=0; +dil per step
    s_bidx[tid] = b;
    s_wout[tid] = (j == dil - 1) ? 1 : 0;
  }
  __syncthreads();

  const int myRowB = s_rowb[M];
  const float*          xraw = (const float*)xin;
  const unsigned short* xbf  = (const unsigned short*)xin;

  for (int t = 0; t < nc; ++t) {
    // ---------------- GI = x_t @ Wih^T + bih ----------------
    if (RAW) {
      if (tid < 16) {
        int rb = s_rowb[tid];
        int b = rb / S, j = rb - b * S;
        int gt = t * dil + j;
        int tt = (flags & FLAG_REV) ? (S - 1 - gt) : gt;
        s_xrow[tid] = xraw[(long long)b * S + tt];
      }
      __syncthreads();
      for (int e = tid; e < 16 * NT16; e += 64) {
        int m = e / NT16, n = e - m * NT16;
        s_ggi[e] = s_xrow[m] * s_wcol[n] + s_bih[n];
      }
    } else {
      const unsigned short* xrow = xbf + (long long)(myRowB + t * dil) * IpadX;
      if (t + 1 < nc)
        __builtin_prefetch(xbf + (long long)(myRowB + (t + 1) * dil) * IpadX, 0, 1);
      // hoist A-fragments (x row) into registers once; reuse over all N-tiles
      FragBF ax[KC_gi];
      #pragma unroll
      for (int kc = 0; kc < KC_gi; ++kc) {
        const int kbA = kc * 32 + hi * 8;
        ax[kc].q[0] = *(const uint4*)(xrow + kbA);
        ax[kc].q[1] = *(const uint4*)(xrow + kbA + 16);
      }
      for (int i = 0; i < NT / 2; ++i) {              // uniform, constant trip count
        const int nt = 2 * i + wv;
        const int n  = nt * 16 + M;
        v8f acc = {};
        #pragma unroll
        for (int kc = 0; kc < KC_gi; ++kc) {
          FragBF Bm;
          const unsigned short* wr = s_wih + (size_t)n * IpadX + kc * 32 + hi * 8;
          Bm.q[0] = *(const uint4*)(wr);
          Bm.q[1] = *(const uint4*)(wr + 16);
          acc = __builtin_amdgcn_wmma_f32_16x16x32_bf16(false, ax[kc].v, false, Bm.v,
                                                        (short)0, acc, false, false);
        }
        float bb = s_bih[n];
        #pragma unroll
        for (int v = 0; v < 8; ++v) s_ggi[(v + hi * 8) * NT16 + n] = acc[v] + bb;
      }
    }

    // ---------------- GH = h @ Whh^T + bhh ----------------
    {
      // hoist A-fragments (h) into registers once; reuse over all N-tiles
      FragBF ah[KC_gh];
      const unsigned short* hr = s_hbf + (size_t)M * Hpad;
      #pragma unroll
      for (int kc = 0; kc < KC_gh; ++kc) {
        const int kbA = kc * 32 + hi * 8;
        ah[kc].q[0] = *(const uint4*)(hr + kbA);
        ah[kc].q[1] = *(const uint4*)(hr + kbA + 16);
      }
      for (int i = 0; i < NT / 2; ++i) {              // uniform, constant trip count
        const int nt = 2 * i + wv;
        const int n  = nt * 16 + M;
        v8f acc = {};
        #pragma unroll
        for (int kc = 0; kc < KC_gh; ++kc) {
          FragBF Bm;
          const unsigned short* wr = s_whh + (size_t)n * Hpad + kc * 32 + hi * 8;
          Bm.q[0] = *(const uint4*)(wr);
          Bm.q[1] = *(const uint4*)(wr + 16);
          acc = __builtin_amdgcn_wmma_f32_16x16x32_bf16(false, ah[kc].v, false, Bm.v,
                                                        (short)0, acc, false, false);
        }
        float bb = s_bhh[n];
        #pragma unroll
        for (int v = 0; v < 8; ++v) s_ggh[(v + hi * 8) * NT16 + n] = acc[v] + bb;
      }
    }
    __syncthreads();

    // ---------------- gates + h update + outputs ----------------
    for (int e = tid; e < 16 * H; e += 64) {
      int m = e / H, col = e - m * H;
      float ir  = s_ggi[m * NT16 + col];
      float iz  = s_ggi[m * NT16 + col + H];
      float in_ = s_ggi[m * NT16 + col + 2 * H];
      float hr  = s_ggh[m * NT16 + col];
      float hz  = s_ggh[m * NT16 + col + H];
      float hn  = s_ggh[m * NT16 + col + 2 * H];
      float r = fast_sigmoid(ir + hr);
      float z = fast_sigmoid(iz + hz);
      float n = fast_tanh(in_ + r * hn);
      float hprev = s_h[m * Hpad + col];
      float hnew  = (1.0f - z) * n + z * hprev;
      s_h[m * Hpad + col]   = hnew;
      unsigned short hb = f32_to_bf16_rne(hnew);
      s_hbf[m * Hpad + col] = hb;
      if (IpadOut > 0)
        yout[(long long)(s_rowb[m] + t * dil) * IpadOut + col] = hb;
      if (t == nc - 1 && s_wout[m])
        dout[(long long)s_bidx[m] * 400 + out_off + col] = hnew;
    }
    __syncthreads();
  }
}

// ---------------------------------------------------------------------------
struct LCfg {
  int threeH, H, Hpad, I, IpadX, IpadOut, NT, KC_gi, KC_gh, dil, nc, tiles, flags, out_off;
};

extern "C" void kernel_launch(void* const* d_in, const int* in_sizes, int n_in,
                              void* d_out, int out_size, void* d_ws, size_t ws_size,
                              hipStream_t stream)
{
  (void)in_sizes; (void)n_in; (void)out_size;
  const int B = 1024, S = 512;

  static const LCfg cfg[6] = {
    //3H    H  Hp    I  IpX IpO  NT kci kch dil  nc  tiles  flags     off
    {300, 100, 128,   1,  32, 128, 20, 1, 4,  1, 512,  64, 0,           0},
    {150,  50,  64, 100, 128,  64, 10, 4, 2,  4, 128, 256, 0,         100},
    {150,  50,  64,  50,  64,   0, 10, 2, 2, 16,  32,  64, FLAG_LAST, 150},
    {300, 100, 128,   1,  32, 128, 20, 1, 4,  1, 512,  64, FLAG_REV,  200},
    {150,  50,  64, 100, 128,  64, 10, 4, 2,  4, 128, 256, 0,         300},
    {150,  50,  64,  50,  64,   0, 10, 2, 2, 16,  32,  64, FLAG_LAST, 350},
  };

  // ---- carve workspace ----
  char* base = (char*)d_ws;
  size_t off = 0;
  auto alloc = [&](size_t bytes) -> void* {
    size_t o = off;
    off = (off + bytes + 255) & ~(size_t)255;
    return (void*)(base + o);
  };
  unsigned short *wihp[6], *whhp[6];
  float *bihp[6], *bhhp[6], *wcolp[6];
  for (int l = 0; l < 6; ++l) {
    int NT16 = cfg[l].NT * 16;
    wihp[l]  = (unsigned short*)alloc((size_t)NT16 * cfg[l].IpadX * 2);
    whhp[l]  = (unsigned short*)alloc((size_t)NT16 * cfg[l].Hpad * 2);
    bihp[l]  = (float*)alloc((size_t)NT16 * 4);
    bhhp[l]  = (float*)alloc((size_t)NT16 * 4);
    wcolp[l] = (float*)alloc((size_t)NT16 * 4);
  }
  unsigned short* Y1 = (unsigned short*)alloc((size_t)B * S * 128 * 2); // layer-1 out (bf16, pad 128)
  unsigned short* Y2 = (unsigned short*)alloc((size_t)B * S * 64 * 2);  // layer-2 out (bf16, pad 64)
  if (off > ws_size) return; // not enough scratch: bail deterministically

  hipFuncSetAttribute(reinterpret_cast<const void*>(&gru_layer<100,128, 32,128,20,1,4,true >),
                      hipFuncAttributeMaxDynamicSharedMemorySize, 160 * 1024);
  hipFuncSetAttribute(reinterpret_cast<const void*>(&gru_layer< 50, 64,128, 64,10,4,2,false>),
                      hipFuncAttributeMaxDynamicSharedMemorySize, 160 * 1024);
  hipFuncSetAttribute(reinterpret_cast<const void*>(&gru_layer< 50, 64, 64,  0,10,2,2,false>),
                      hipFuncAttributeMaxDynamicSharedMemorySize, 160 * 1024);

  // ---- prep weights (f32 -> padded bf16) ----
  for (int l = 0; l < 6; ++l) {
    const float* wih = (const float*)d_in[1 + 4 * l + 0];
    const float* whh = (const float*)d_in[1 + 4 * l + 1];
    const float* bih = (const float*)d_in[1 + 4 * l + 2];
    const float* bhh = (const float*)d_in[1 + 4 * l + 3];
    int NT16 = cfg[l].NT * 16;
    prep_weights_kernel<<<64, 256, 0, stream>>>(
        wih, whh, bih, bhh, wihp[l], whhp[l], bihp[l], bhhp[l], wcolp[l],
        cfg[l].threeH, cfg[l].I, cfg[l].IpadX, cfg[l].H, cfg[l].Hpad, NT16);
  }
  // zero activation buffers so K/N padding lanes stay inert
  zero_bytes_kernel<<<2048, 256, 0, stream>>>((uint4*)Y1, (long long)B * S * 128 * 2 / 16);
  zero_bytes_kernel<<<1024, 256, 0, stream>>>((uint4*)Y2, (long long)B * S * 64 * 2 / 16);

  const float* data = (const float*)d_in[0];
  float* out = (float*)d_out;

  const void*     xs[6] = { data, Y1, Y2, data, Y1, Y2 };
  unsigned short* ys[6] = { Y1,   Y2, nullptr, Y1, Y2, nullptr };

  for (int l = 0; l < 6; ++l) {
    const LCfg& c = cfg[l];
    int NT16 = c.NT * 16;
    size_t lds = (size_t)16 * 4 * 4                       // rowb/bidx/wout/xrow
               + (size_t)NT16 * 4 * 3                     // bih/bhh/wcol
               + (size_t)16 * NT16 * 4 * 2                // ggi/ggh
               + (size_t)16 * c.Hpad * 4                  // h (f32)
               + (size_t)16 * c.Hpad * 2                  // h (bf16)
               + (size_t)NT16 * c.Hpad * 2                // Whh bf16
               + (size_t)NT16 * c.IpadX * 2;              // Wih bf16
    switch (l % 3) {
      case 0:
        gru_layer<100,128, 32,128,20,1,4,true ><<<c.tiles, 64, lds, stream>>>(
            xs[l], ys[l], wihp[l], whhp[l], bihp[l], bhhp[l], wcolp[l],
            out, c.out_off, S, c.dil, c.nc, c.flags);
        break;
      case 1:
        gru_layer< 50, 64,128, 64,10,4,2,false><<<c.tiles, 64, lds, stream>>>(
            xs[l], ys[l], wihp[l], whhp[l], bihp[l], bhhp[l], wcolp[l],
            out, c.out_off, S, c.dil, c.nc, c.flags);
        break;
      default:
        gru_layer< 50, 64, 64,  0,10,2,2,false><<<c.tiles, 64, lds, stream>>>(
            xs[l], ys[l], wihp[l], whhp[l], bihp[l], bhhp[l], wcolp[l],
            out, c.out_off, S, c.dil, c.nc, c.flags);
        break;
    }
  }
}
